// Encoder_14370960572729
// MI455X (gfx1250) — matrix operand order repaired and checked by reading, not compiled, not run on hardware
//
#include <hip/hip_runtime.h>
#include <math.h>

// ---------------- problem constants ----------------
#define B_SZ   2048
#define C_SZ   128
#define F_SZ   128
#define NKEEP  16

// ---------------- workspace layout (bytes, 256-aligned) ----------------
#define OFF_POS     0u           // pos[F*C] fp32                     (65536 B)
#define OFF_SS      65536u       // scale/shift stages [6][2][128]    (6144 B)
#define OFF_STATS   71680u       // BN stats [5][2][128]              (5120 B)
#define OFF_WSPEC   76800u       // packed bf16 w_spec                (32768 B)
#define OFF_WPOSO   109568u      // packed bf16 w_pos_out             (32768 B)
#define OFF_WATOMS  142336u      // packed bf16 w_atoms 128x256       (65536 B)
#define OFF_CWS     207872u      // per-layer BN-folded packed conv W (98304 B)
#define OFF_BEFF    306176u      // per-layer effective bias [128]    (512 B)
#define OFF_ACTA    306688u      // activation ping (bf16)            (67108864 B)
#define OFF_ACTB    67415552u    // activation pong (bf16)            (67108864 B)
#define OFF_HFIN    134524416u   // final h fp32                      (134217728 B)

// output offsets in floats: (amps, posn, atoms, verb) concatenated flat
#define O_AMPS  0u
#define O_POSN  32768u
#define O_ATOMS 4227072u
#define O_VERB  12615680u

// ---------------- types / helpers ----------------
typedef __attribute__((ext_vector_type(16))) __bf16        v16bf;
typedef __attribute__((ext_vector_type(8)))  float         v8f;
typedef __attribute__((ext_vector_type(8)))  unsigned int  v8u;
typedef __attribute__((ext_vector_type(4)))  unsigned int  v4u;
typedef __attribute__((ext_vector_type(4)))  unsigned int  u32x4;
typedef __attribute__((ext_vector_type(8)))  int           i32x8;
typedef __attribute__((ext_vector_type(4)))  int           i32x4;

#if defined(__has_builtin)
#  if __has_builtin(__builtin_amdgcn_tensor_load_to_lds)
#    define HAVE_TDM 1
#  endif
#endif
#ifndef HAVE_TDM
#  define HAVE_TDM 0
#endif

__device__ __forceinline__ unsigned short f2bf(float f) {
  unsigned int u = __float_as_uint(f);
  u += 0x7fffu + ((u >> 16) & 1u);          // round-to-nearest-even
  return (unsigned short)(u >> 16);
}
__device__ __forceinline__ float bf2f(unsigned short h) {
  return __uint_as_float(((unsigned int)h) << 16);
}

// A fragment, 16x32 bf16 (contiguous k order): lane l holds M=l%16;
// halves 0..15 = K kb..kb+15, kb = kk + (l/16)*16
__device__ __forceinline__ v16bf load_afrag(const unsigned short* rowp, int kbase) {
  v4u lo = *(const v4u*)(rowp + kbase);
  v4u hi = *(const v4u*)(rowp + kbase + 8);
  v8u t;
  t[0]=lo[0]; t[1]=lo[1]; t[2]=lo[2]; t[3]=lo[3];
  t[4]=hi[0]; t[5]=hi[1]; t[6]=hi[2]; t[7]=hi[3];
  return __builtin_bit_cast(v16bf, t);
}
// B fragment from pre-packed weights: tile (kt,nt) = 512 contiguous halves;
// lane l reads halves [l*16, l*16+16)  (one 32B load)
__device__ __forceinline__ v16bf load_bfrag(const unsigned short* mat, int ktiles,
                                            int kk, int nt, int lane) {
  const int tile = nt * ktiles + (kk >> 5);
  v8u t = *(const v8u*)(mat + tile * 512 + lane * 16);
  return __builtin_bit_cast(v16bf, t);
}
__device__ __forceinline__ v8f wmma_bf16(v16bf a, v16bf b, v8f c) {
  return __builtin_amdgcn_wmma_f32_16x16x32_bf16(false, a, false, b, (short)0, c,
                                                 false, false);
}

// ---------------- kernel 1: positional embedding  pos[f][c] ----------------
__global__ void k_pos(const float* __restrict__ w_pos, const float* __restrict__ b_pos,
                      float* __restrict__ pos) {
  int tid = blockIdx.x * 256 + threadIdx.x;     // 16384
  int f = tid >> 7, c = tid & 127;
  float p = -1.f + 2.f * (float)f / 127.f;
  float acc = b_pos[c] + p * w_pos[c];
  float s = 1.f;
  for (int i = 0; i < 16; ++i) {
    float sp = p * s;
    acc += __sinf(sp) * w_pos[(1 + i) * C_SZ + c] +
           __cosf(sp) * w_pos[(17 + i) * C_SZ + c];
    s *= 2.f;
  }
  pos[tid] = acc;
}

// ---------------- kernel 2: pack static weights to bf16 WMMA-B layout -------
__global__ void k_pack(const float* __restrict__ w_spec, const float* __restrict__ w_poso,
                       const float* __restrict__ w_atoms,
                       unsigned short* __restrict__ wspec_pk,
                       unsigned short* __restrict__ wposo_pk,
                       unsigned short* __restrict__ watoms_pk) {
  int tid = blockIdx.x * 256 + threadIdx.x;     // 65536 total
  if (tid < 16384) {                            // w_spec [k=ci][n=co], N=128
    int pi = tid, tile = pi >> 9, lane = (pi >> 4) & 31, e = pi & 15;
    int kt = tile & 3, nt = tile >> 2;
    int k = kt * 32 + (lane >> 4) * 16 + e, n = nt * 16 + (lane & 15);
    wspec_pk[pi] = f2bf(w_spec[k * 128 + n]);
  } else if (tid < 32768) {                     // w_pos_out [k=c][n], N=128
    int pi = tid - 16384;
    int tile = pi >> 9, lane = (pi >> 4) & 31, e = pi & 15;
    int kt = tile & 3, nt = tile >> 2;
    int k = kt * 32 + (lane >> 4) * 16 + e, n = nt * 16 + (lane & 15);
    wposo_pk[pi] = f2bf(w_poso[k * 128 + n]);
  } else {                                      // w_atoms [k=c][n], N=256
    int pi = tid - 32768;
    int tile = pi >> 9, lane = (pi >> 4) & 31, e = pi & 15;
    int kt = tile & 3, nt = tile >> 2;          // Ktiles=4, Ntiles=16
    int k = kt * 32 + (lane >> 4) * 16 + e, n = nt * 16 + (lane & 15);
    watoms_pk[pi] = f2bf(w_atoms[k * 256 + n]);
  }
}

// ---------------- kernel 3: zero BN stats, identity stage-0 scale/shift -----
__global__ void k_init(float* __restrict__ ss, float* __restrict__ stats) {
  int t = threadIdx.x;
  if (t < 128) { ss[t] = 1.f; ss[128 + t] = 0.f; }
  for (int i = t; i < 1280; i += 256) stats[i] = 0.f;
}

// -------- kernel 4: fold input-BN into conv weights (per layer) -------------
// W'[co,ci,t] = W*scale[ci] (packed bf16);  b'[co] = b[co] + sum W*shift[ci]
__global__ void k_scalew(const float* __restrict__ conv_w, const float* __restrict__ convb,
                         const float* __restrict__ ssin, unsigned short* __restrict__ cw_s,
                         float* __restrict__ beff, int l) {
  const int blk = blockIdx.x, t = threadIdx.x;
  if (blk < 192) {
    int r = blk * 256 + t;                      // 0..49151 = 3 packed matrices
    int tp = r >> 14, pi = r & 16383;
    int tile = pi >> 9, lane = (pi >> 4) & 31, e = pi & 15;
    int kt = tile & 3, nt = tile >> 2;
    int k = kt * 32 + (lane >> 4) * 16 + e, n = nt * 16 + (lane & 15); // k=ci,n=co
    float w = conv_w[((l * 128 + n) * 128 + k) * 3 + tp];
    cw_s[r] = f2bf(w * ssin[k]);
  } else if (t < 128) {
    float acc = convb[t];
    for (int ci = 0; ci < 128; ++ci) {
      float sh = ssin[128 + ci];
      for (int tp = 0; tp < 3; ++tp)
        acc += conv_w[((l * 128 + t) * 128 + ci) * 3 + tp] * sh;
    }
    beff[t] = acc;
  }
}

// ---------------- kernel 5: h0 = x^T @ w_spec + b_spec + pos  (bf16 out) ----
__global__ void __launch_bounds__(256) k_embed(const float* __restrict__ x,
    const unsigned short* __restrict__ wpk, const float* __restrict__ b_spec,
    const float* __restrict__ pos, unsigned short* __restrict__ act) {
  __shared__ __attribute__((aligned(16))) unsigned short Ash[128 * 136];
  const int b = blockIdx.x, t = threadIdx.x;
  const float* xb = x + (size_t)b * 16384;
  for (int idx = t; idx < 16384; idx += 256) {  // x[b][ci][f] -> A[f][ci] bf16
    int ci = idx >> 7, f = idx & 127;
    Ash[f * 136 + ci] = f2bf(xb[idx]);
  }
  __builtin_prefetch(wpk + (t & 31) * 512, 0, 3);   // global_prefetch_b8
  __syncthreads();
  const int wave = t >> 5, lane = t & 31, m0 = wave * 16;
  v8f acc[8] = {};
  const unsigned short* arow = Ash + (m0 + (lane & 15)) * 136;
  const int kb = (lane >> 4) * 16;
#pragma unroll
  for (int kk = 0; kk < 128; kk += 32) {
    v16bf a = load_afrag(arow, kk + kb);
#pragma unroll
    for (int nt = 0; nt < 8; ++nt) {
      v16bf bm = load_bfrag(wpk, 4, kk, nt, lane);
      acc[nt] = wmma_bf16(a, bm, acc[nt]);
    }
  }
  unsigned short* ob = act + (size_t)b * 16384;
  const int fr0 = m0 + 8 * (lane >> 4);
#pragma unroll
  for (int nt = 0; nt < 8; ++nt) {
    int n = nt * 16 + (lane & 15);
    float bias = b_spec[n];
#pragma unroll
    for (int r = 0; r < 8; ++r) {
      int f = fr0 + r;
      ob[f * 128 + n] = f2bf(acc[nt][r] + bias + pos[f * 128 + n]);
    }
  }
}

// ------- kernel 6: dilated conv (BN folded in W) + leaky + BN-stats ---------
// Activation tile staged by the Tensor Data Mover with hardware LDS padding:
// 128 rows of 256B + 16B pad -> 272B row stride (matches 136-half layout).
__global__ void __launch_bounds__(256) k_conv(const unsigned short* __restrict__ in,
    unsigned short* __restrict__ out, const unsigned short* __restrict__ wpk,
    const float* __restrict__ beff, float* __restrict__ stats, int dil) {
  __shared__ __attribute__((aligned(16))) unsigned short Ash[182 * 136]; // halo 27
  __shared__ float ssum[128], ssq[128];
  const int b = blockIdx.x, t = threadIdx.x;
  if (t < 128) { ssum[t] = 0.f; ssq[t] = 0.f; }
  // zero halo rows [0,27) and [155,182)  (54 rows * 68 dwords)
  for (int idx = t; idx < 54 * 68; idx += 256) {
    int rr = idx / 68, wq = idx - rr * 68;
    int row = (rr < 27) ? rr : rr + 128;
    ((unsigned int*)Ash)[row * 68 + wq] = 0u;
  }
  const unsigned short* ib = in + (size_t)b * 16384;
#if HAVE_TDM
  if (t < 32) {                                 // one wave issues the DMA
    unsigned long long ga = (unsigned long long)(const void*)ib;
    unsigned int lds0 = (unsigned int)(unsigned long long)(const void*)Ash + 27u * 272u;
    // D# group0: count=1 | lds_addr | global_addr[56:0] | type=2
    u32x4 g0 = { 1u, lds0, (unsigned int)ga,
                 (unsigned int)((ga >> 32) & 0x01FFFFFFu) | 0x80000000u };
    // D# group1: data_size=2B, pad_enable, pad_interval=64DW, pad_amount=4DW,
    // tensor_dim0=128, tensor_dim1=128, tile_dim0=128, tile_dim1=128, stride0=128
    i32x8 g1 = { (int)((1u << 16) | (1u << 20) | (5u << 22) | (3u << 25)),
                 (int)(128u << 16),   // tensor_dim0[15:0] at bits 63:48
                 (int)(128u << 16),   // tensor_dim1[15:0] at bits 95:80
                 (int)(128u << 16),   // tile_dim0 at bits 127:112
                 128,                 // tile_dim1
                 128,                 // tensor_dim0_stride[31:0]
                 0, 0 };
    i32x4 g2 = { 0, 0, 0, 0 }, g3 = { 0, 0, 0, 0 };
    i32x8 g4 = { 0, 0, 0, 0, 0, 0, 0, 0 };
    __builtin_amdgcn_tensor_load_to_lds(g0, g1, g2, g3, g4, 0);
    __builtin_amdgcn_s_wait_tensorcnt((short)0);
  }
#else
  for (int idx = t; idx < 128 * 64; idx += 256) {  // raw bf16 copy fallback
    int f = idx >> 6, wq = idx & 63;
    ((unsigned int*)Ash)[(f + 27) * 68 + wq] = ((const unsigned int*)ib)[f * 64 + wq];
  }
#endif
  __builtin_prefetch(wpk + t * 384, 0, 3);      // warm 3 scaled weight matrices
  __syncthreads();
  const int wave = t >> 5, lane = t & 31, m0 = wave * 16;
  const int kb = (lane >> 4) * 16;
  const int mrow = m0 + (lane & 15) + 27;
  v8f acc[8] = {};
  for (int tap = 0; tap < 3; ++tap) {
    const unsigned short* arow = Ash + (mrow + (tap - 1) * dil) * 136;
    const unsigned short* wm = wpk + tap * 16384;
#pragma unroll
    for (int kk = 0; kk < 128; kk += 32) {
      v16bf a = load_afrag(arow, kk + kb);
#pragma unroll
      for (int nt = 0; nt < 8; ++nt) {
        v16bf bm = load_bfrag(wm, 4, kk, nt, lane);
        acc[nt] = wmma_bf16(a, bm, acc[nt]);
      }
    }
  }
  unsigned short* ob = out + (size_t)b * 16384;
  const int fr0 = m0 + 8 * (lane >> 4);
#pragma unroll
  for (int nt = 0; nt < 8; ++nt) {
    int ch = nt * 16 + (lane & 15);
    float bias = beff[ch];
    float s = 0.f, q = 0.f;
#pragma unroll
    for (int r = 0; r < 8; ++r) {
      float v = acc[nt][r] + bias;
      v = v > 0.f ? v : 0.2f * v;               // leaky_relu 0.2
      ob[(fr0 + r) * 128 + ch] = f2bf(v);
      s += v; q += v * v;
    }
    atomicAdd(&ssum[ch], s);
    atomicAdd(&ssq[ch], q);
  }
  __syncthreads();
  if (t < 128) {
    atomicAdd(&stats[t], ssum[t]);
    atomicAdd(&stats[128 + t], ssq[t]);
  }
}

// ---------------- kernel 7: BN stats -> per-channel scale/shift -------------
__global__ void k_bnfin(const float* __restrict__ stats, const float* __restrict__ g,
                        const float* __restrict__ bb, float* __restrict__ ssout) {
  int c = threadIdx.x;
  const float inv_n = 1.f / (float)(B_SZ * F_SZ);
  float mu = stats[c] * inv_n;
  float var = stats[128 + c] * inv_n - mu * mu;
  float sc = g[c] * rsqrtf(var + 1e-5f);
  ssout[c] = sc;
  ssout[128 + c] = bb[c] - mu * sc;
}

// ---------------- kernel 8: h_final = BN4(act) + pos  (fp32) ----------------
__global__ void __launch_bounds__(256) k_finalh(const unsigned short* __restrict__ act,
    const float* __restrict__ ss, const float* __restrict__ pos, float* __restrict__ h) {
  const int b = blockIdx.x, t = threadIdx.x;
  const unsigned short* ib = act + (size_t)b * 16384;
  float* ob = h + (size_t)b * 16384;
  for (int idx = t; idx < 16384; idx += 256) {
    int c = idx & 127;
    ob[idx] = ss[c] * bf2f(ib[idx]) + ss[128 + c] + pos[idx];
  }
}

// ---------------- kernel 9: attention, top-k, heads (WMMA posn/atoms) -------
__global__ void __launch_bounds__(256) k_head(const float* __restrict__ h,
    const float* __restrict__ w_attn, const float* __restrict__ b_attn,
    const float* __restrict__ w_verb, const float* __restrict__ b_verb,
    const float* __restrict__ w_amps, const float* __restrict__ b_amps,
    const unsigned short* __restrict__ wposo_pk, const float* __restrict__ b_poso,
    const unsigned short* __restrict__ watoms_pk, const float* __restrict__ b_atoms,
    float* __restrict__ out) {
  __shared__ float lg[128], wat[128], aggS[128];
  __shared__ float selF[16][128];
  __shared__ __attribute__((aligned(16))) unsigned short selB[16 * 136];
  __shared__ float vals[16];
  __shared__ int idxs[16];
  const int b = blockIdx.x, t = threadIdx.x;
  const float* hb = h + (size_t)b * 16384;
  if (t < 128) wat[t] = w_attn[t];
  __syncthreads();
  if (t < 128) {
    float d = b_attn[0];
    for (int c = 0; c < 128; ++c) d += hb[t * 128 + c] * wat[c];
    lg[t] = d;
  }
  __syncthreads();
  if (t == 0) {                                 // softmax + serial top-16
    float mx = lg[0];
    for (int f = 1; f < 128; ++f) mx = fmaxf(mx, lg[f]);
    float sm = 0.f;
    for (int f = 0; f < 128; ++f) { float e = __expf(lg[f] - mx); lg[f] = e; sm += e; }
    float inv = 1.f / sm;
    for (int f = 0; f < 128; ++f) lg[f] *= inv;
    for (int j = 0; j < NKEEP; ++j) {
      int am = 0; float av = lg[0];
      for (int f = 1; f < 128; ++f) if (lg[f] > av) { av = lg[f]; am = f; }
      vals[j] = av; idxs[j] = am; lg[am] = -1.f;
    }
  }
  __syncthreads();
  for (int idx = t; idx < 16 * 128; idx += 256) { // sel = gather * attn value
    int j = idx >> 7, c = idx & 127;
    float v = hb[idxs[j] * 128 + c] * vals[j];
    selF[j][c] = v;
    selB[j * 136 + c] = f2bf(v);
  }
  __syncthreads();
  if (t < 128) {
    float a = 0.f;
    for (int j = 0; j < NKEEP; ++j) a += selF[j][t];
    aggS[t] = a;
  }
  __syncthreads();
  if (t < 128) {                                // verb = agg@W + b + agg
    float v = b_verb[t] + aggS[t];
    for (int k = 0; k < 128; ++k) v += aggS[k] * w_verb[k * 128 + t];
    out[O_VERB + (size_t)b * 128 + t] = v;
  }
  if (t >= 128 && t < 144) {                    // amps = (sel@w_amps+b)^2
    int j = t - 128;
    float a = b_amps[0];
    for (int c = 0; c < 128; ++c) a += selF[j][c] * w_amps[c];
    out[O_AMPS + (size_t)b * 16 + j] = a * a;
  }
  __syncthreads();
  const int wave = t >> 5, lane = t & 31;
  const int kb = (lane >> 4) * 16;
  const unsigned short* arow = selB + (lane & 15) * 136;
  {
    v8f acc = {};
#pragma unroll
    for (int kk = 0; kk < 128; kk += 32) {
      v16bf a = load_afrag(arow, kk + kb);
      v16bf bm = load_bfrag(wposo_pk, 4, kk, wave, lane);
      acc = wmma_bf16(a, bm, acc);
    }
    int n = wave * 16 + (lane & 15);
    float bias = b_poso[n];
    int j0 = 8 * (lane >> 4);
#pragma unroll
    for (int r = 0; r < 8; ++r)
      out[O_POSN + ((size_t)b * 16 + j0 + r) * 128 + n] = acc[r] + bias;
  }
#pragma unroll
  for (int g = 0; g < 2; ++g) {
    int nt = wave + g * 8;
    v8f acc = {};
#pragma unroll
    for (int kk = 0; kk < 128; kk += 32) {
      v16bf a = load_afrag(arow, kk + kb);
      v16bf bm = load_bfrag(watoms_pk, 4, kk, nt, lane);
      acc = wmma_bf16(a, bm, acc);
    }
    int n = nt * 16 + (lane & 15);
    float bias = b_atoms[n];
    int j0 = 8 * (lane >> 4);
#pragma unroll
    for (int r = 0; r < 8; ++r)
      out[O_ATOMS + ((size_t)b * 16 + j0 + r) * 256 + n] = acc[r] + bias;
  }
}

// ---------------- host launcher ----------------
extern "C" void kernel_launch(void* const* d_in, const int* in_sizes, int n_in,
                              void* d_out, int out_size, void* d_ws, size_t ws_size,
                              hipStream_t stream) {
  (void)in_sizes; (void)n_in; (void)out_size; (void)ws_size;
  const float* x       = (const float*)d_in[0];
  const float* w_spec  = (const float*)d_in[1];
  const float* b_spec  = (const float*)d_in[2];
  const float* w_pos   = (const float*)d_in[3];
  const float* b_pos   = (const float*)d_in[4];
  const float* conv_w  = (const float*)d_in[5];
  const float* conv_b  = (const float*)d_in[6];
  const float* bn_g    = (const float*)d_in[7];
  const float* bn_b    = (const float*)d_in[8];
  const float* w_attn  = (const float*)d_in[9];
  const float* b_attn  = (const float*)d_in[10];
  const float* w_verb  = (const float*)d_in[11];
  const float* b_verb  = (const float*)d_in[12];
  const float* w_amps  = (const float*)d_in[13];
  const float* b_amps  = (const float*)d_in[14];
  const float* w_poso  = (const float*)d_in[15];
  const float* b_poso  = (const float*)d_in[16];
  const float* w_atoms = (const float*)d_in[17];
  const float* b_atoms = (const float*)d_in[18];
  float* out = (float*)d_out;
  char* ws = (char*)d_ws;
  float* pos   = (float*)(ws + OFF_POS);
  float* ss    = (float*)(ws + OFF_SS);
  float* stats = (float*)(ws + OFF_STATS);
  unsigned short* wspec_pk  = (unsigned short*)(ws + OFF_WSPEC);
  unsigned short* wposo_pk  = (unsigned short*)(ws + OFF_WPOSO);
  unsigned short* watoms_pk = (unsigned short*)(ws + OFF_WATOMS);
  unsigned short* cw_s      = (unsigned short*)(ws + OFF_CWS);
  float* beff = (float*)(ws + OFF_BEFF);
  unsigned short* actA = (unsigned short*)(ws + OFF_ACTA);
  unsigned short* actB = (unsigned short*)(ws + OFF_ACTB);
  float* hfin = (float*)(ws + OFF_HFIN);

  k_pos <<<64, 256, 0, stream>>>(w_pos, b_pos, pos);
  k_pack<<<256, 256, 0, stream>>>(w_spec, w_poso, w_atoms,
                                  wspec_pk, wposo_pk, watoms_pk);
  k_init<<<1, 256, 0, stream>>>(ss, stats);
  k_embed<<<B_SZ, 256, 0, stream>>>(x, wspec_pk, b_spec, pos, actA);

  const int dil[5] = {1, 3, 9, 27, 1};
  unsigned short* bufs[2] = {actA, actB};
  for (int l = 0; l < 5; ++l) {
    unsigned short* ain  = bufs[l & 1];
    unsigned short* aout = bufs[(l & 1) ^ 1];
    k_scalew<<<193, 256, 0, stream>>>(conv_w, conv_b + l * 128, ss + l * 256,
                                      cw_s, beff, l);
    k_conv  <<<B_SZ, 256, 0, stream>>>(ain, aout, cw_s, beff,
                                       stats + l * 256, dil[l]);
    k_bnfin <<<1, 128, 0, stream>>>(stats + l * 256, bn_g + l * 128,
                                    bn_b + l * 128, ss + (l + 1) * 256);
  }
  k_finalh<<<B_SZ, 256, 0, stream>>>(bufs[1], ss + 5 * 256, pos, hfin);
  k_head  <<<B_SZ, 256, 0, stream>>>(hfin, w_attn, b_attn, w_verb, b_verb,
                                     w_amps, b_amps, wposo_pk, b_poso,
                                     watoms_pk, b_atoms, out);
}
